// NeuralTuringMachine_25718264168582
// MI455X (gfx1250) — matrix operand order repaired and checked by reading, not compiled, not run on hardware
//
#include <hip/hip_runtime.h>
#include <math.h>

#define I_SZ  256
#define H_SZ  512
#define M_SZ  16384
#define D_SZ  256
#define T_LEN 128
#define G_SZ  2048   // 4*H

typedef __attribute__((ext_vector_type(16))) _Float16 v16h;
typedef __attribute__((ext_vector_type(8)))  float    v8f;

__device__ __forceinline__ float wsum(float v) {
#pragma unroll
  for (int o = 16; o > 0; o >>= 1) v += __shfl_xor(v, o, 32);
  return v;
}
__device__ __forceinline__ float sigm(float x) { return 1.f / (1.f + expf(-x)); }
__device__ __forceinline__ float softplus(float x) {
  return (x > 20.f) ? x : log1pf(expf(x));
}

// ---------------------------------------------------------------------------
// init: copy memory0 -> mem, compute row norms, copy rw/ww, zero h/c
// ---------------------------------------------------------------------------
__global__ void k_init(const float* __restrict__ memory0,
                       const float* __restrict__ read_w0,
                       const float* __restrict__ write_w0,
                       float* mem, float* memN, float* rw, float* ww,
                       float* hst, float* cst) {
  int b = blockIdx.x, tid = threadIdx.x, lane = tid & 31, wave = tid >> 5;
  for (int r = 0; r < 32; ++r) {
    int m = b * 256 + wave * 32 + r;
    const float* src = memory0 + (size_t)m * D_SZ;
    float* dst = mem + (size_t)m * D_SZ;
    float ss = 0.f;
    for (int j = lane; j < D_SZ; j += 32) { float v = src[j]; dst[j] = v; ss += v * v; }
    ss = wsum(ss);
    if (lane == 0) memN[m] = fmaxf(sqrtf(ss), 1e-12f);
  }
  int idx = b * 256 + tid;
  rw[idx] = read_w0[idx];
  ww[idx] = write_w0[idx];
  if (b == 0 && tid < 256) {
    hst[tid] = 0.f; hst[tid + 256] = 0.f;
    cst[tid] = 0.f; cst[tid + 256] = 0.f;
  }
}

// ---------------------------------------------------------------------------
// read_vec partials: partial[b][d] = sum over this block's 256 rows of rw*mem
// ---------------------------------------------------------------------------
__global__ void k_read_partial(const float* __restrict__ rw,
                               const float* __restrict__ mem,
                               float* __restrict__ partial) {
  int b = blockIdx.x, d = threadIdx.x;
  const float* base = mem + (size_t)b * 256 * D_SZ;
  float acc = 0.f;
  for (int r = 0; r < 256; ++r)
    acc += rw[b * 256 + r] * base[(size_t)r * D_SZ + d];
  partial[b * D_SZ + d] = acc;
}

// ---------------------------------------------------------------------------
// reduce partials -> read_vec, build ci = [x_t, read_vec]
// ---------------------------------------------------------------------------
__global__ void k_ci(const float* __restrict__ partial,
                     const float* __restrict__ inputs, int t,
                     float* __restrict__ rvec, float* __restrict__ ci) {
  int d = threadIdx.x;
  float s = 0.f;
  for (int b = 0; b < 64; ++b) s += partial[b * D_SZ + d];
  rvec[d] = s;
  ci[d] = inputs[(size_t)t * I_SZ + d];
  ci[I_SZ + d] = s;
}

// ---------------------------------------------------------------------------
// gates = ci @ W_ih.T + b_ih + h @ W_hh.T + b_hh  (wave per gate row)
// ---------------------------------------------------------------------------
__global__ void k_gates(const float* __restrict__ ci, const float* __restrict__ h,
                        const float* __restrict__ W_ih, const float* __restrict__ W_hh,
                        const float* __restrict__ b_ih, const float* __restrict__ b_hh,
                        float* __restrict__ gates) {
  int tid = threadIdx.x, lane = tid & 31, wave = tid >> 5;
  for (int rr = 0; rr < 8; ++rr) {
    int row = blockIdx.x * 64 + wave * 8 + rr;
    const float* wi = W_ih + (size_t)row * H_SZ;
    const float* wh = W_hh + (size_t)row * H_SZ;
    float a = 0.f;
    for (int k = lane; k < H_SZ; k += 32) a += ci[k] * wi[k] + h[k] * wh[k];
    a = wsum(a);
    if (lane == 0) gates[row] = a + b_ih[row] + b_hh[row];
  }
}

// ---------------------------------------------------------------------------
// LSTM state update + all head projections + output row  (single block, 512)
// hscal: [0]=beta_w [1]=g_w [2]=gamma_w [3..5]=s_w  [6]=beta_r [7]=g_r
//        [8]=gamma_r [9..11]=s_r
// ---------------------------------------------------------------------------
__global__ void k_lstm_heads(
    const float* __restrict__ gates, float* h, float* c,
    const float* __restrict__ rvec,
    const float* __restrict__ Wo, const float* __restrict__ bo,
    const float* wWk, const float* wbk, const float* wWb, const float* wbb,
    const float* wWg, const float* wbg, const float* wWs, const float* wbs,
    const float* wWp, const float* wbp,
    const float* wWe, const float* wbe, const float* wWa, const float* wba,
    const float* rWk, const float* rbk, const float* rWb, const float* rbb,
    const float* rWg, const float* rbg, const float* rWs, const float* rbs,
    const float* rWp, const float* rbp,
    float* knw, float* erasev, float* addv, float* knr, float* hscal,
    float* out_t) {
  __shared__ float co[H_SZ];
  __shared__ float keyw_s[D_SZ];
  __shared__ float keyr_s[D_SZ];
  __shared__ float small_s[12];
  __shared__ float red[512];
  __shared__ float nw2, nr2;
  int tid = threadIdx.x, lane = tid & 31, wave = tid >> 5;

  // LSTM cell
  {
    float iv = gates[tid], fv = gates[H_SZ + tid];
    float gv = gates[2 * H_SZ + tid], ov = gates[3 * H_SZ + tid];
    float cn = sigm(fv) * c[tid] + sigm(iv) * tanhf(gv);
    float hn = sigm(ov) * tanhf(cn);
    c[tid] = cn; h[tid] = hn; co[tid] = hn;
  }
  __syncthreads();

  auto wdot = [&](const float* Wr) {
    float a = 0.f;
    for (int k = lane; k < H_SZ; k += 32) a += co[k] * Wr[k];
    return wsum(a);
  };

  // keys (both heads), erase, add : 16 waves x 16 rows
  for (int r = wave; r < D_SZ; r += 16) {
    float kwv = wdot(wWk + (size_t)r * H_SZ);
    if (lane == 0) keyw_s[r] = kwv + wbk[r];
    float krv = wdot(rWk + (size_t)r * H_SZ);
    if (lane == 0) keyr_s[r] = krv + rbk[r];
    float ev = wdot(wWe + (size_t)r * H_SZ);
    if (lane == 0) erasev[r] = sigm(ev + wbe[r]);
    float av = wdot(wWa + (size_t)r * H_SZ);
    if (lane == 0) addv[r] = av + wba[r];
  }
  // 12 small projections, one per wave
  {
    const float* rp = nullptr;
    if      (wave == 0)  rp = wWb;
    else if (wave == 1)  rp = wWg;
    else if (wave == 2)  rp = wWs;
    else if (wave == 3)  rp = wWs + H_SZ;
    else if (wave == 4)  rp = wWs + 2 * H_SZ;
    else if (wave == 5)  rp = wWp;
    else if (wave == 6)  rp = rWb;
    else if (wave == 7)  rp = rWg;
    else if (wave == 8)  rp = rWs;
    else if (wave == 9)  rp = rWs + H_SZ;
    else if (wave == 10) rp = rWs + 2 * H_SZ;
    else if (wave == 11) rp = rWp;
    if (rp) { float v = wdot(rp); if (lane == 0) small_s[wave] = v; }
  }
  __syncthreads();

  if (tid == 0) {
    // write head scalars
    hscal[0] = softplus(small_s[0] + wbb[0]);
    hscal[1] = sigm(small_s[1] + wbg[0]);
    hscal[2] = softplus(small_s[5] + wbp[0]) + 1.f;
    float a0 = small_s[2] + wbs[0], a1 = small_s[3] + wbs[1], a2 = small_s[4] + wbs[2];
    float mx = fmaxf(a0, fmaxf(a1, a2));
    float e0 = expf(a0 - mx), e1 = expf(a1 - mx), e2 = expf(a2 - mx);
    float si = 1.f / (e0 + e1 + e2);
    hscal[3] = e0 * si; hscal[4] = e1 * si; hscal[5] = e2 * si;
    // read head scalars
    hscal[6] = softplus(small_s[6] + rbb[0]);
    hscal[7] = sigm(small_s[7] + rbg[0]);
    hscal[8] = softplus(small_s[11] + rbp[0]) + 1.f;
    a0 = small_s[8] + rbs[0]; a1 = small_s[9] + rbs[1]; a2 = small_s[10] + rbs[2];
    mx = fmaxf(a0, fmaxf(a1, a2));
    e0 = expf(a0 - mx); e1 = expf(a1 - mx); e2 = expf(a2 - mx);
    si = 1.f / (e0 + e1 + e2);
    hscal[9] = e0 * si; hscal[10] = e1 * si; hscal[11] = e2 * si;
  }

  // key norms + normalized keys
  float vsq = (tid < D_SZ) ? keyw_s[tid] * keyw_s[tid] : 0.f;
  red[tid] = vsq; __syncthreads();
  for (int s = 256; s > 0; s >>= 1) { if (tid < s) red[tid] += red[tid + s]; __syncthreads(); }
  if (tid == 0) nw2 = red[0];
  __syncthreads();
  vsq = (tid < D_SZ) ? keyr_s[tid] * keyr_s[tid] : 0.f;
  red[tid] = vsq; __syncthreads();
  for (int s = 256; s > 0; s >>= 1) { if (tid < s) red[tid] += red[tid + s]; __syncthreads(); }
  if (tid == 0) nr2 = red[0];
  __syncthreads();
  if (tid < D_SZ) {
    knw[tid] = keyw_s[tid] / fmaxf(sqrtf(nw2), 1e-12f);
    knr[tid] = keyr_s[tid] / fmaxf(sqrtf(nr2), 1e-12f);
  }
  __syncthreads();

  // output row: out = [co, read_vec_old] @ Wo.T + bo
  for (int r = wave; r < I_SZ; r += 16) {
    const float* wr = Wo + (size_t)r * (H_SZ + D_SZ);
    float a = 0.f;
    for (int j = lane; j < H_SZ + D_SZ; j += 32) {
      float xv = (j < H_SZ) ? co[j] : rvec[j - H_SZ];
      a += xv * wr[j];
    }
    a = wsum(a);
    if (lane == 0) out_t[r] = a + bo[r];
  }
}

// ---------------------------------------------------------------------------
// content score via WMMA: logits[m] = beta * (kn . mem[m]) / max(||mem[m]||,1e-12)
// A tile = 16 mem rows x 32 K (f16) loaded as 4x b128 per chunk;
// B fragments (kn in column N=0, lanes 0/16) precomputed once, branch-free.
// ---------------------------------------------------------------------------
__global__ void k_score(const float* __restrict__ mem,
                        const float* __restrict__ memN,
                        const float* __restrict__ kn,
                        const float* __restrict__ betap,
                        float* __restrict__ logits) {
  int tid = threadIdx.x, lane = tid & 31, wave = tid >> 5;
  int g = blockIdx.x * 8 + wave;            // 512 waves total
  bool hi = lane >= 16;
  bool l0 = (lane == 0), l16 = (lane == 16);

  // Precompute all 8 B fragments: loads are unconditional (uniform addresses),
  // per-lane selection is a v_cndmask, no control flow.
  v16h bf[8];
#pragma unroll
  for (int kc = 0; kc < 8; ++kc) {
    int kb = kc * 32;
#pragma unroll
    for (int e = 0; e < 16; ++e) {
      float k0 = kn[kb + (e < 8 ? e : e + 8)];
      float k1 = kn[kb + 8 + (e < 8 ? e : e + 8)];
      float bv = l0 ? k0 : (l16 ? k1 : 0.f);
      bf[kc][e] = (_Float16)bv;
    }
  }
  float beta = betap[0];

#pragma unroll
  for (int tt = 0; tt < 2; ++tt) {
    int tile = g + tt * 512;                // 1024 tiles of 16 rows
    int r0 = tile * 16;
    int row = r0 + (lane & 15);
    const float* mrow = mem + (size_t)row * D_SZ;
    if (tt == 0)
      __builtin_prefetch(mem + (size_t)(row + 512 * 16) * D_SZ, 0, 0);
    v8f acc = {};
#pragma unroll
    for (int kc = 0; kc < 8; ++kc) {
      int kb = kc * 32;
      // lanes<16: e0..7 = K kb+0..7,  e8..15 = K kb+16..23
      // lanes>=16: e0..7 = K kb+8..15, e8..15 = K kb+24..31
      const float* pa = mrow + kb + (hi ? 8 : 0);
      float4 x0 = *reinterpret_cast<const float4*>(pa);
      float4 x1 = *reinterpret_cast<const float4*>(pa + 4);
      float4 y0 = *reinterpret_cast<const float4*>(pa + 16);
      float4 y1 = *reinterpret_cast<const float4*>(pa + 20);
      v16h a;
      a[0]  = (_Float16)x0.x; a[1]  = (_Float16)x0.y;
      a[2]  = (_Float16)x0.z; a[3]  = (_Float16)x0.w;
      a[4]  = (_Float16)x1.x; a[5]  = (_Float16)x1.y;
      a[6]  = (_Float16)x1.z; a[7]  = (_Float16)x1.w;
      a[8]  = (_Float16)y0.x; a[9]  = (_Float16)y0.y;
      a[10] = (_Float16)y0.z; a[11] = (_Float16)y0.w;
      a[12] = (_Float16)y1.x; a[13] = (_Float16)y1.y;
      a[14] = (_Float16)y1.z; a[15] = (_Float16)y1.w;
      acc = __builtin_amdgcn_wmma_f32_16x16x32_f16(
          false, a, false, bf[kc], (short)0, acc, false, false);
    }
    // C layout: N=0 lives in lanes 0 (rows r0..r0+7) and 16 (rows r0+8..r0+15)
    if (lane == 0) {
#pragma unroll
      for (int j = 0; j < 8; ++j) { int m = r0 + j; logits[m] = beta * acc[j] / memN[m]; }
    } else if (lane == 16) {
#pragma unroll
      for (int j = 0; j < 8; ++j) { int m = r0 + 8 + j; logits[m] = beta * acc[j] / memN[m]; }
    }
  }
}

// ---------------------------------------------------------------------------
// full addressing pipeline over M: softmax -> interpolate -> shift -> sharpen
// single block of 1024, 16 elements per thread, deterministic tree reductions
// ---------------------------------------------------------------------------
__global__ void k_attend(const float* __restrict__ logits,
                         const float* __restrict__ prev_w,
                         const float* __restrict__ gp, const float* __restrict__ sp,
                         const float* __restrict__ gammap,
                         float* __restrict__ out_w,
                         float* __restrict__ bufA, float* __restrict__ bufB) {
  __shared__ float red[1024];
  __shared__ float sh[2];
  int tid = threadIdx.x;

  float mx = -1e30f;
  for (int i = 0; i < 16; ++i) mx = fmaxf(mx, logits[tid + i * 1024]);
  red[tid] = mx; __syncthreads();
  for (int s = 512; s > 0; s >>= 1) { if (tid < s) red[tid] = fmaxf(red[tid], red[tid + s]); __syncthreads(); }
  if (tid == 0) sh[0] = red[0];
  __syncthreads();
  float gmax = sh[0];

  float ssum = 0.f;
  for (int i = 0; i < 16; ++i) {
    int m = tid + i * 1024;
    float e = expf(logits[m] - gmax);
    bufA[m] = e; ssum += e;
  }
  red[tid] = ssum; __syncthreads();
  for (int s = 512; s > 0; s >>= 1) { if (tid < s) red[tid] += red[tid + s]; __syncthreads(); }
  if (tid == 0) sh[1] = 1.f / red[0];
  __syncthreads();
  float invS = sh[1];
  float gg = gp[0];

  for (int i = 0; i < 16; ++i) {
    int m = tid + i * 1024;
    float cw = bufA[m] * invS;
    bufA[m] = gg * cw + (1.f - gg) * prev_w[m];
  }
  __syncthreads();

  float s0 = sp[0], s1 = sp[1], s2 = sp[2], gamma = gammap[0];
  float psum = 0.f;
  for (int i = 0; i < 16; ++i) {
    int m = tid + i * 1024;
    int mp = (m + 1) & (M_SZ - 1);
    int mm = (m - 1) & (M_SZ - 1);
    float sv = s0 * bufA[mp] + s1 * bufA[m] + s2 * bufA[mm];
    float wv = powf(sv, gamma);
    bufB[m] = wv; psum += wv;
  }
  red[tid] = psum; __syncthreads();
  for (int s = 512; s > 0; s >>= 1) { if (tid < s) red[tid] += red[tid + s]; __syncthreads(); }
  if (tid == 0) sh[1] = 1.f / (red[0] + 1e-8f);
  __syncthreads();
  float inv2 = sh[1];
  for (int i = 0; i < 16; ++i) {
    int m = tid + i * 1024;
    out_w[m] = bufB[m] * inv2;
  }
}

// ---------------------------------------------------------------------------
// mem = mem*(1 - ww x erase) + ww x add ; refresh row norms in the same pass
// ---------------------------------------------------------------------------
__global__ void k_memupdate(float* __restrict__ mem, float* __restrict__ memN,
                            const float* __restrict__ ww,
                            const float* __restrict__ erasev,
                            const float* __restrict__ addv) {
  int b = blockIdx.x, tid = threadIdx.x, lane = tid & 31, wave = tid >> 5;
  for (int r = 0; r < 32; ++r) {
    int m = b * 256 + wave * 32 + r;
    float wv = ww[m];
    float* row = mem + (size_t)m * D_SZ;
    float ss = 0.f;
    for (int j = lane; j < D_SZ; j += 32) {
      float v = row[j];
      v = v * (1.f - wv * erasev[j]) + wv * addv[j];
      row[j] = v; ss += v * v;
    }
    ss = wsum(ss);
    if (lane == 0) memN[m] = fmaxf(sqrtf(ss), 1e-12f);
  }
}

// ---------------------------------------------------------------------------
extern "C" void kernel_launch(void* const* d_in, const int* in_sizes, int n_in,
                              void* d_out, int out_size, void* d_ws, size_t ws_size,
                              hipStream_t stream) {
  const float* inputs  = (const float*)d_in[0];
  const float* memory0 = (const float*)d_in[1];
  const float* read_w0 = (const float*)d_in[2];
  const float* write_w0= (const float*)d_in[3];
  const float* W_ih    = (const float*)d_in[4];
  const float* W_hh    = (const float*)d_in[5];
  const float* b_ih    = (const float*)d_in[6];
  const float* b_hh    = (const float*)d_in[7];
  const float* Wo      = (const float*)d_in[8];
  const float* bo      = (const float*)d_in[9];
  const float* rWk = (const float*)d_in[10]; const float* rbk = (const float*)d_in[11];
  const float* rWb = (const float*)d_in[12]; const float* rbb = (const float*)d_in[13];
  const float* rWg = (const float*)d_in[14]; const float* rbg = (const float*)d_in[15];
  const float* rWs = (const float*)d_in[16]; const float* rbs = (const float*)d_in[17];
  const float* rWp = (const float*)d_in[18]; const float* rbp = (const float*)d_in[19];
  const float* wWk = (const float*)d_in[20]; const float* wbk = (const float*)d_in[21];
  const float* wWb = (const float*)d_in[22]; const float* wbb = (const float*)d_in[23];
  const float* wWg = (const float*)d_in[24]; const float* wbg = (const float*)d_in[25];
  const float* wWs = (const float*)d_in[26]; const float* wbs = (const float*)d_in[27];
  const float* wWp = (const float*)d_in[28]; const float* wbp = (const float*)d_in[29];
  const float* wWe = (const float*)d_in[30]; const float* wbe = (const float*)d_in[31];
  const float* wWa = (const float*)d_in[32]; const float* wba = (const float*)d_in[33];
  float* out = (float*)d_out;

  // workspace layout (~17.3 MB of f32)
  float* W      = (float*)d_ws;
  float* mem    = W;                                   // M*D
  float* memN   = mem    + (size_t)M_SZ * D_SZ;        // M
  float* rw     = memN   + M_SZ;                       // M
  float* ww     = rw     + M_SZ;                       // M
  float* hst    = ww     + M_SZ;                       // H
  float* cst    = hst    + H_SZ;                       // H
  float* rvec   = cst    + H_SZ;                       // D
  float* ci     = rvec   + D_SZ;                       // I+D
  float* gates  = ci     + (I_SZ + D_SZ);              // 4H
  float* knw    = gates  + G_SZ;                       // D
  float* erasev = knw    + D_SZ;                       // D
  float* addv   = erasev + D_SZ;                       // D
  float* knr    = addv   + D_SZ;                       // D
  float* hscal  = knr    + D_SZ;                       // 16
  float* logits = hscal  + 16;                         // M
  float* bufA   = logits + M_SZ;                       // M
  float* bufB   = bufA   + M_SZ;                       // M
  float* partial= bufB   + M_SZ;                       // 64*D

  k_init<<<64, 256, 0, stream>>>(memory0, read_w0, write_w0,
                                 mem, memN, rw, ww, hst, cst);

  for (int t = 0; t < T_LEN; ++t) {
    k_read_partial<<<64, 256, 0, stream>>>(rw, mem, partial);
    k_ci<<<1, 256, 0, stream>>>(partial, inputs, t, rvec, ci);
    k_gates<<<32, 256, 0, stream>>>(ci, hst, W_ih, W_hh, b_ih, b_hh, gates);
    k_lstm_heads<<<1, 512, 0, stream>>>(
        gates, hst, cst, rvec, Wo, bo,
        wWk, wbk, wWb, wbb, wWg, wbg, wWs, wbs, wWp, wbp,
        wWe, wbe, wWa, wba,
        rWk, rbk, rWb, rbb, rWg, rbg, rWs, rbs, rWp, rbp,
        knw, erasev, addv, knr, hscal, out + (size_t)t * I_SZ);
    // write-head addressing on old mem
    k_score<<<64, 256, 0, stream>>>(mem, memN, knw, hscal + 0, logits);
    k_attend<<<1, 1024, 0, stream>>>(logits, ww, hscal + 1, hscal + 3, hscal + 2,
                                     ww, bufA, bufB);
    // memory write + fresh norms
    k_memupdate<<<64, 256, 0, stream>>>(mem, memN, ww, erasev, addv);
    // read-head addressing on new mem
    k_score<<<64, 256, 0, stream>>>(mem, memN, knr, hscal + 6, logits);
    k_attend<<<1, 1024, 0, stream>>>(logits, rw, hscal + 7, hscal + 9, hscal + 8,
                                     rw, bufA, bufB);
  }
}
